// Attention_v5_21603685499616
// MI455X (gfx1250) — matrix operand order repaired and checked by reading, not compiled, not run on hardware
//
#include <hip/hip_runtime.h>

// ---------------------------------------------------------------------------
// MI455X (gfx1250) attention: f16 WMMA GEMMs + flash-style attention.
//  - Weights converted/transposed to f16 once; x converted to f16 once.
//  - GEMMs: 128x128x32 tiles, double-buffered LDS filled with
//    global_load_async_to_lds_b128 (ASYNCcnt), 8 waves x 8 WMMA per K-step.
//  - Attention: per-(n,head) workgroup, online softmax, WMMA for S and P@V.
// ---------------------------------------------------------------------------

typedef __attribute__((ext_vector_type(4)))  float    v4f;
typedef __attribute__((ext_vector_type(4)))  _Float16 v4h;
typedef __attribute__((ext_vector_type(8)))  _Float16 v8h;
typedef __attribute__((ext_vector_type(16))) _Float16 v16h;
typedef __attribute__((ext_vector_type(8)))  float    v8f;

// A-matrix fragment (16x32 f16): lane half 0 holds K {0..7,16..23}, half 1
// holds K {8..15,24..31}; row = lane&15  (ISA 7.12.2, 16-bit A layout).
__device__ __forceinline__ v16h load_a_frag(const _Float16* base, long stride) {
  const int lane = threadIdx.x & 31;
  const _Float16* p = base + (long)(lane & 15) * stride + ((lane >> 4) << 3);
  v8h lo = *(const v8h*)p;
  v8h hi = *(const v8h*)(p + 16);
  return __builtin_shufflevector(lo, hi, 0,1,2,3,4,5,6,7,8,9,10,11,12,13,14,15);
}

// B-matrix fragment (32x16 f16): lane = column, half 0 holds K {0..7,16..23},
// half 1 holds K {8..15,24..31}. base points at a [col][k] tile.
__device__ __forceinline__ v16h load_b_frag(const _Float16* base, long stride) {
  const int lane = threadIdx.x & 31;
  const _Float16* p = base + (long)(lane & 15) * stride + ((lane >> 4) << 3);
  v8h lo = *(const v8h*)p;
  v8h hi = *(const v8h*)(p + 16);
  return __builtin_shufflevector(lo, hi, 0,1,2,3,4,5,6,7,8,9,10,11,12,13,14,15);
}

__device__ __forceinline__ v8f wmma_f16(v16h a, v16h b, v8f c) {
  return __builtin_amdgcn_wmma_f32_16x16x32_f16(false, a, false, b,
                                                (short)0, c, false, false);
}

// 16B global -> LDS async copy (ASYNCcnt-tracked). LDS byte address is the
// low 32 bits of the flat shared pointer (ISA 10.2 aperture mapping).
__device__ __forceinline__ void async_copy_b128(const _Float16* g,
                                                const _Float16* l) {
  asm volatile("global_load_async_to_lds_b128 %0, %1, off"
               :: "v"((unsigned)(unsigned long long)l),
                  "v"((unsigned long long)g)
               : "memory");
}

// ---------------------------------------------------------------------------
// Prep kernels (run once per launch; all operands resident in L2 afterwards).
// ---------------------------------------------------------------------------
__global__ __launch_bounds__(256) void cvt_f32_to_f16_kernel(
    const float* __restrict__ src, _Float16* __restrict__ dst, long n) {
  const long i = ((long)blockIdx.x * 256 + threadIdx.x) * 4;
  if (i < n) {
    v4f v = *(const v4f*)(src + i);
    v4h h;
#pragma unroll
    for (int x = 0; x < 4; ++x) h[x] = (_Float16)v[x];
    *(v4h*)(dst + i) = h;
  }
}

// w[K][N] f32 -> wt[N][K] f16 (one block per output row n).
__global__ __launch_bounds__(256) void transpose_cvt_kernel(
    const float* __restrict__ w, _Float16* __restrict__ wt, int K, int N) {
  const int n = blockIdx.x;
  for (int k = threadIdx.x; k < K; k += 256)
    wt[(long)n * K + k] = (_Float16)w[(long)k * N + n];
}

// ---------------------------------------------------------------------------
// Tiled GEMM: C[M,N] = A16[M,K] @ Bt[N,K]^T + bias.   (both inputs f16)
// MODE 0: write f16, scale cols with (col%192)<128 by `scale` (QKV epilogue).
// MODE 1: write f32 (final projection).
// 256 threads (8 waves). Tile 128x128x32; wave sub-tile 32x64.
// Double-buffered LDS, async global->LDS staging, prefetch depth 1.
// ---------------------------------------------------------------------------
template <int MODE>
__global__ __launch_bounds__(256) void gemm_wmma_kernel(
    const _Float16* __restrict__ A16, const _Float16* __restrict__ Bt,
    const float* __restrict__ bias, void* __restrict__ Cp,
    int M, int N, int K, float scale) {
  __shared__ __align__(16) _Float16 As[2][128][40];  // [m][k], pad 32->40
  __shared__ __align__(16) _Float16 Bs[2][128][40];  // [n][k]

  const int tileM = blockIdx.y * 128;
  const int tileN = blockIdx.x * 128;
  const int t     = threadIdx.x;
  const int w     = t >> 5;
  const int lane  = t & 31;
  const int waveM = (w >> 1) * 32;   // 4 waves along M
  const int waveN = (w & 1) * 64;    // 2 waves along N
  const int lane15 = lane & 15;
  const int half   = lane >> 4;

  v8f acc[2][4] = {};

  // Stage one 128x32 A tile + 128x32 B tile: 512 x 16B chunks each matrix
  // half; thread t copies chunks t and t+256 => 4 async instrs per wave.
  auto stage = [&](int it) {
    const int buf = it & 1;
    const int k0  = it << 5;
#pragma unroll
    for (int s = 0; s < 2; ++s) {
      const int c    = t + s * 256;
      const int row  = c >> 2;
      const int coff = (c & 3) * 8;
      async_copy_b128(A16 + (long)(tileM + row) * K + k0 + coff,
                      &As[buf][row][coff]);
      async_copy_b128(Bt + (long)(tileN + row) * K + k0 + coff,
                      &Bs[buf][row][coff]);
    }
  };

  const int nIter = K >> 5;
  stage(0);
  for (int it = 0; it < nIter; ++it) {
    const int cur = it & 1;
    if (it + 1 < nIter) {
      stage(it + 1);  // prefetch next tile into the other buffer
      // async loads complete in order per wave: <=4 left => tile `it` landed
      asm volatile("s_wait_asynccnt 0x4" ::: "memory");
    } else {
      asm volatile("s_wait_asynccnt 0x0" ::: "memory");
    }
    __syncthreads();

    v16h a0 = load_a_frag(&As[cur][waveM + 0][0], 40);
    v16h a1 = load_a_frag(&As[cur][waveM + 16][0], 40);
#pragma unroll
    for (int jf = 0; jf < 4; ++jf) {
      v16h b = load_b_frag(&Bs[cur][waveN + jf * 16][0], 40);
      acc[0][jf] = wmma_f16(a0, b, acc[0][jf]);
      acc[1][jf] = wmma_f16(a1, b, acc[1][jf]);
    }
    __syncthreads();  // reads done before this buffer is refilled
  }

  // ---- epilogue: VGPR e -> row e (half0) / 8+e (half1), col = lane15 ----
#pragma unroll
  for (int i = 0; i < 2; ++i) {
#pragma unroll
    for (int jf = 0; jf < 4; ++jf) {
      const int col = tileN + waveN + jf * 16 + lane15;
      const float bb = bias[col];
      const bool doScale = (MODE == 0) && ((col % 192) < 128);
#pragma unroll
      for (int e = 0; e < 8; ++e) {
        const int row = tileM + waveM + i * 16 + half * 8 + e;
        float v = acc[i][jf][e] + bb;
        if (MODE == 0) {
          if (doScale) v *= scale;
          ((_Float16*)Cp)[(long)row * N + col] = (_Float16)v;
        } else {
          ((float*)Cp)[(long)row * N + col] = v;
        }
      }
    }
  }
}

// ---------------------------------------------------------------------------
// Flash-style attention over the 'a' axis (256 tokens, dh = 64).
// Grid: (h=8, n=256). 8 waves; wave w owns query rows [w*32, w*32+32).
// qkv16 layout per token: [h*192 + {q:0..63 | k:64..127 | v:128..191}],
// q and k already scaled by dh^-0.25.
// ---------------------------------------------------------------------------
__global__ __launch_bounds__(256) void attn_wmma_kernel(
    const _Float16* __restrict__ qkv, _Float16* __restrict__ o16) {
  __shared__ __align__(16) _Float16 Vt[64][264];     // V transposed [d][j]
  __shared__ __align__(16) _Float16 Ps[8][32][72];   // per-wave P scratch

  const int h       = blockIdx.x;
  const int tokBase = blockIdx.y * 256;
  const int t    = threadIdx.x;
  const int w    = t >> 5;
  const int lane = t & 31;
  const int lane15 = lane & 15;
  const int half   = lane >> 4;

  // ---- stage V transposed into LDS (thread t loads key row t) ----
  {
    const _Float16* vp = qkv + (long)(tokBase + t) * 1536 + h * 192 + 128;
#pragma unroll
    for (int d = 0; d < 64; ++d) Vt[d][t] = vp[d];
  }
  __syncthreads();

  // ---- Q fragments straight from global (row-major in d) ----
  v16h qf[2][2];
  {
    const _Float16* qb = qkv + (long)(tokBase + w * 32) * 1536 + h * 192;
#pragma unroll
    for (int i = 0; i < 2; ++i)
#pragma unroll
      for (int ks = 0; ks < 2; ++ks)
        qf[i][ks] = load_a_frag(qb + (long)(i * 16) * 1536 + ks * 32, 1536);
  }

  float mrow[2][8], lrow[2][8];
#pragma unroll
  for (int i = 0; i < 2; ++i)
#pragma unroll
    for (int e = 0; e < 8; ++e) { mrow[i][e] = -1e30f; lrow[i][e] = 0.f; }
  v8f oacc[2][4] = {};

  for (int kb = 0; kb < 4; ++kb) {
    // ---- S block = Q @ K^T  (K frags from global: lane = key row j) ----
    v8f sacc[2][4] = {};
#pragma unroll
    for (int ks = 0; ks < 2; ++ks) {
#pragma unroll
      for (int jf = 0; jf < 4; ++jf) {
        v16h kf = load_b_frag(
            qkv + (long)(tokBase + kb * 64 + jf * 16) * 1536 + h * 192 + 64 +
                ks * 32,
            1536);
        sacc[0][jf] = wmma_f16(qf[0][ks], kf, sacc[0][jf]);
        sacc[1][jf] = wmma_f16(qf[1][ks], kf, sacc[1][jf]);
      }
    }

    // ---- online softmax: row r lives at (i, e, half); cols across 16 lanes --
#pragma unroll
    for (int i = 0; i < 2; ++i) {
#pragma unroll
      for (int e = 0; e < 8; ++e) {
        float bm = sacc[i][0][e];
#pragma unroll
        for (int jf = 1; jf < 4; ++jf) bm = fmaxf(bm, sacc[i][jf][e]);
#pragma unroll
        for (int m = 1; m < 16; m <<= 1) bm = fmaxf(bm, __shfl_xor(bm, m, 32));
        const float mnew = fmaxf(mrow[i][e], bm);
        const float cf   = __expf(mrow[i][e] - mnew);
        float rs = 0.f;
#pragma unroll
        for (int jf = 0; jf < 4; ++jf) {
          const float p = __expf(sacc[i][jf][e] - mnew);
          sacc[i][jf][e] = p;
          rs += p;
        }
#pragma unroll
        for (int m = 1; m < 16; m <<= 1) rs += __shfl_xor(rs, m, 32);
        lrow[i][e] = lrow[i][e] * cf + rs;
        mrow[i][e] = mnew;
#pragma unroll
        for (int nd = 0; nd < 4; ++nd) oacc[i][nd][e] *= cf;
      }
    }

    // ---- write P (f16) to wave-private LDS scratch for C->A relayout ----
#pragma unroll
    for (int i = 0; i < 2; ++i)
#pragma unroll
      for (int jf = 0; jf < 4; ++jf)
#pragma unroll
        for (int e = 0; e < 8; ++e)
          Ps[w][i * 16 + half * 8 + e][jf * 16 + lane15] =
              (_Float16)sacc[i][jf][e];
    // DS ops are per-wave in-order; drain the split DS counter before reads.
    asm volatile("s_wait_dscnt 0" ::: "memory");

    // ---- O += P @ V ----
#pragma unroll
    for (int ks = 0; ks < 2; ++ks) {
      v16h a0 = load_a_frag(&Ps[w][0][ks * 32], 72);
      v16h a1 = load_a_frag(&Ps[w][16][ks * 32], 72);
#pragma unroll
      for (int nd = 0; nd < 4; ++nd) {
        v16h bf = load_b_frag(&Vt[nd * 16][kb * 64 + ks * 32], 264);
        oacc[0][nd] = wmma_f16(a0, bf, oacc[0][nd]);
        oacc[1][nd] = wmma_f16(a1, bf, oacc[1][nd]);
      }
    }
  }

  // ---- normalize and write O (f16) as [token][h*64 + d] ----
#pragma unroll
  for (int i = 0; i < 2; ++i) {
#pragma unroll
    for (int e = 0; e < 8; ++e) {
      const float inv = 1.f / lrow[i][e];
      const int row = tokBase + w * 32 + i * 16 + half * 8 + e;
#pragma unroll
      for (int nd = 0; nd < 4; ++nd) {
        const int col = h * 64 + nd * 16 + lane15;
        o16[(long)row * 512 + col] = (_Float16)(oacc[i][nd][e] * inv);
      }
    }
  }
}

// ---------------------------------------------------------------------------
extern "C" void kernel_launch(void* const* d_in, const int* in_sizes, int n_in,
                              void* d_out, int out_size, void* d_ws,
                              size_t ws_size, hipStream_t stream) {
  (void)in_sizes; (void)n_in; (void)out_size; (void)ws_size;
  const float* x      = (const float*)d_in[0];  // [65536, 512]
  const float* w_qkv  = (const float*)d_in[1];  // [512, 1536]
  const float* b_qkv  = (const float*)d_in[2];  // [1536]
  const float* w_proj = (const float*)d_in[3];  // [512, 512]
  const float* b_proj = (const float*)d_in[4];  // [512]
  float* out = (float*)d_out;                   // [65536, 512]

  const int M = 65536, Kd = 512, Nqkv = 1536, Nproj = 512;
  const float scale = 0.35355339059327373f;  // 64^-0.25

  char* ws = (char*)d_ws;
  size_t off = 0;
  _Float16* x16    = (_Float16*)(ws + off); off += (size_t)M * Kd * 2;      // 64 MB
  _Float16* qkv16  = (_Float16*)(ws + off); off += (size_t)M * Nqkv * 2;    // 192 MB
  _Float16* o16    = (_Float16*)(ws + off); off += (size_t)M * Nproj * 2;   // 64 MB
  _Float16* wqkvT  = (_Float16*)(ws + off); off += (size_t)Nqkv * Kd * 2;   // 1.5 MB
  _Float16* wprojT = (_Float16*)(ws + off);                                 // 0.5 MB

  // 0) one-time converts: x -> f16; weights -> f16, transposed to [N][K].
  {
    const long n = (long)M * Kd;
    cvt_f32_to_f16_kernel<<<(unsigned)(n / 4 / 256), 256, 0, stream>>>(x, x16, n);
  }
  transpose_cvt_kernel<<<Nqkv, 256, 0, stream>>>(w_qkv, wqkvT, Kd, Nqkv);
  transpose_cvt_kernel<<<Nproj, 256, 0, stream>>>(w_proj, wprojT, Kd, Nproj);

  // 1) QKV projection -> f16 workspace, q/k pre-scaled.
  gemm_wmma_kernel<0><<<dim3(Nqkv / 128, M / 128), 256, 0, stream>>>(
      x16, wqkvT, b_qkv, qkv16, M, Nqkv, Kd, scale);

  // 2) Attention per (n, head).
  attn_wmma_kernel<<<dim3(8, 256), 256, 0, stream>>>(qkv16, o16);

  // 3) Output projection -> f32 result.
  gemm_wmma_kernel<1><<<dim3(Nproj / 128, M / 128), 256, 0, stream>>>(
      o16, wprojT, b_proj, out, M, Nproj, Kd, 1.0f);
}
